// CodecTransformerLayer_75935021793779
// MI455X (gfx1250) — matrix operand order repaired and checked by reading, not compiled, not run on hardware
//
#include <hip/hip_runtime.h>

#define DIM     1024
#define SEQ     2048
#define BATCH   2
#define ROWS    (BATCH * SEQ)
#define NH      16
#define HD      64
#define HIDDEN  4096
#define WINDOW  512

typedef __attribute__((ext_vector_type(16))) _Float16 v16h;
typedef __attribute__((ext_vector_type(8)))  float    v8f;
typedef __attribute__((ext_vector_type(4)))  unsigned int u32x4;
typedef __attribute__((ext_vector_type(8)))  int      i32x8;
typedef __attribute__((ext_vector_type(4)))  int      i32x4;

#if __has_builtin(__builtin_amdgcn_tensor_load_to_lds)
#define USE_TDM_B 1
#else
#define USE_TDM_B 0
#endif

union Frag16 { v16h v; uint4 u[2]; };

// ---------------------------------------------------------------- utilities

__global__ void k_cvt_f16(const float* __restrict__ src, _Float16* __restrict__ dst, int n) {
    int i = blockIdx.x * blockDim.x + threadIdx.x;
    int stride = gridDim.x * blockDim.x;
    for (; i < n; i += stride) dst[i] = (_Float16)src[i];
}

__device__ inline float2 block_sum2(float a, float b, float* shm /*>=16 floats*/) {
    #pragma unroll
    for (int off = 16; off; off >>= 1) {
        a += __shfl_xor(a, off, 32);
        b += __shfl_xor(b, off, 32);
    }
    int wid = threadIdx.x >> 5, lane = threadIdx.x & 31;
    if (lane == 0) { shm[wid] = a; shm[8 + wid] = b; }
    __syncthreads();
    if (threadIdx.x < 32) {
        float x = (lane < 8) ? shm[lane] : 0.f;
        float y = (lane < 8) ? shm[8 + lane] : 0.f;
        #pragma unroll
        for (int off = 4; off; off >>= 1) {
            x += __shfl_xor(x, off, 32);
            y += __shfl_xor(y, off, 32);
        }
        if (lane == 0) { shm[0] = x; shm[8] = y; }
    }
    __syncthreads();
    float r1 = shm[0], r2 = shm[8];
    __syncthreads();
    return make_float2(r1, r2);
}

__global__ void __launch_bounds__(256)
k_layernorm_f16(const float* __restrict__ in, const float* __restrict__ gamma,
                _Float16* __restrict__ out, int cols, float eps) {
    __shared__ float shm[16];
    int row = blockIdx.x;
    const float* x = in + (size_t)row * cols;
    float s = 0.f, ss = 0.f;
    for (int c = threadIdx.x; c < cols; c += blockDim.x) { float v = x[c]; s += v; ss += v * v; }
    float2 r = block_sum2(s, ss, shm);
    float mean = r.x / (float)cols;
    float inv  = rsqrtf(r.y / (float)cols - mean * mean + eps);
    for (int c = threadIdx.x; c < cols; c += blockDim.x)
        out[(size_t)row * cols + c] = (_Float16)((x[c] - mean) * inv * gamma[c]);
}

// In-place Q-norm (1024 cols, eps 1e-6) + K-norm (512 cols) on qkv rows [q|k|v]
__global__ void __launch_bounds__(256)
k_qk_norm(_Float16* __restrict__ qkv, const float* __restrict__ qw, const float* __restrict__ kw) {
    __shared__ float shm[16];
    int row = blockIdx.x;
    _Float16* base = qkv + (size_t)row * 2048;
    float s = 0.f, ss = 0.f;
    for (int c = threadIdx.x; c < 1024; c += 256) { float v = (float)base[c]; s += v; ss += v * v; }
    float2 r = block_sum2(s, ss, shm);
    float mean = r.x * (1.f / 1024.f);
    float inv  = rsqrtf(r.y * (1.f / 1024.f) - mean * mean + 1e-6f);
    for (int c = threadIdx.x; c < 1024; c += 256) {
        float v = (float)base[c];
        base[c] = (_Float16)((v - mean) * inv * qw[c]);
    }
    __syncthreads();
    s = 0.f; ss = 0.f;
    for (int c = threadIdx.x; c < 512; c += 256) { float v = (float)base[1024 + c]; s += v; ss += v * v; }
    r = block_sum2(s, ss, shm);
    mean = r.x * (1.f / 512.f);
    inv  = rsqrtf(r.y * (1.f / 512.f) - mean * mean + 1e-6f);
    for (int c = threadIdx.x; c < 512; c += 256) {
        float v = (float)base[1024 + c];
        base[1024 + c] = (_Float16)((v - mean) * inv * kw[c]);
    }
}

// -------------------------------------------------------------- WMMA GEMM
// C[m,n] = sum_k A[m,k] * Bw[n,k]   (A: MxK, Bw: NxK, both K-contiguous)
// Block 256 thr = 8 waves; block tile 128x128; wave tile 32x64 (8 WMMAs/step).
// A tile: global_load_async_to_lds_b128 (ASYNCcnt).  B tile: TDM tensor_load_to_lds
// (TENSORcnt, wave 0, LDS row padding via pad_interval/pad_amount) when available,
// else async path.  Double-buffered LDS.
// MODE 1: store f16.  MODE 2: outF = resid + C*scale[n].  MODE 3: outH = f16(silu(C)*gate)
template <int MODE>
__global__ void __launch_bounds__(256)
k_gemm(const _Float16* __restrict__ A, const _Float16* __restrict__ Bw,
       int M, int N, int K,
       float* __restrict__ outF, _Float16* __restrict__ outH,
       const float* __restrict__ resid, const float* __restrict__ scale,
       const _Float16* __restrict__ gate) {
    (void)M;
    __shared__ _Float16 As[2][128][40];
    __shared__ _Float16 Bs[2][128][40];
    const int tid  = threadIdx.x;
    const int lane = tid & 31;
    const int wid  = tid >> 5;
    const int wm   = wid & 3;   // 0..3 -> 32-row slice of 128
    const int wn   = wid >> 2;  // 0..1 -> 64-col slice of 128
    const int m0   = blockIdx.y * 128;
    const int n0   = blockIdx.x * 128;

    v8f acc[2][4] = {};

    // stage one 128x32 tile-pair into LDS buffer `buf`
    auto issue = [&](int buf, int k0) {
        #pragma unroll
        for (int ch = 0; ch < 2; ++ch) {
            int c   = tid + ch * 256;          // 0..511
            int row = c >> 2;
            int off = (c & 3) * 8;
            unsigned la = (unsigned)(size_t)&As[buf][row][off];
            unsigned long long ga =
                (unsigned long long)(size_t)(A + (size_t)(m0 + row) * K + k0 + off);
            asm volatile("global_load_async_to_lds_b128 %0, %1, off"
                         :: "v"(la), "v"(ga) : "memory");
#if !USE_TDM_B
            unsigned lb = (unsigned)(size_t)&Bs[buf][row][off];
            unsigned long long gb =
                (unsigned long long)(size_t)(Bw + (size_t)(n0 + row) * K + k0 + off);
            asm volatile("global_load_async_to_lds_b128 %0, %1, off"
                         :: "v"(lb), "v"(gb) : "memory");
#endif
        }
#if USE_TDM_B
        if (wid == 0) {
            // Tensor DMA descriptor (ISA 8.3/8.4): 2D tile 32(K) x 128(N), f16,
            // LDS pad: 4 DWORDs after every 16 DWORDs -> 40-half row stride.
            unsigned long long gb =
                (unsigned long long)(size_t)(Bw + (size_t)n0 * K + k0);
            u32x4 g0;
            g0.x = 1u;                                            // count=1 (valid D#)
            g0.y = (unsigned)(size_t)&Bs[buf][0][0];              // lds_addr
            g0.z = (unsigned)gb;                                  // global_addr[31:0]
            g0.w = (unsigned)((gb >> 32) & 0x01FFFFFFull) | 0x80000000u; // addr[56:32], type=2
            i32x8 g1;
            g1[0] = (int)((1u << 16) | (1u << 20) | (3u << 22) | (3u << 25));
            // data_size=2B, pad_enable, pad_interval=16dw, pad_amount=4dw
            g1[1] = (int)(((unsigned)K & 0xFFFFu) << 16);         // tensor_dim0[15:0]
            g1[2] = (int)((((unsigned)K >> 16) & 0xFFFFu) |
                          (((unsigned)N & 0xFFFFu) << 16));       // dim0[31:16] | dim1[15:0]
            g1[3] = (int)((((unsigned)N >> 16) & 0xFFFFu) |
                          (32u << 16));                           // dim1[31:16] | tile_dim0=32
            g1[4] = 128;                                          // tile_dim1=128, tile_dim2=0
            g1[5] = K;                                            // tensor_dim0_stride[31:0]
            g1[6] = 0;
            g1[7] = 0;
            i32x4 z4 = (i32x4)0;
#if defined(__clang_major__) && __clang_major__ >= 23
            i32x8 z8 = (i32x8)0;
            __builtin_amdgcn_tensor_load_to_lds(g0, g1, z4, z4, z8, 0);
#else
            __builtin_amdgcn_tensor_load_to_lds(g0, g1, z4, z4, 0);
#endif
        }
#endif
    };

    issue(0, 0);
    const int half = lane >> 4;
    for (int k0 = 0; k0 < K; k0 += 32) {
        const int buf = (k0 >> 5) & 1;
        if (k0 + 32 < K) {
            issue(buf ^ 1, k0 + 32);
#if USE_TDM_B
            asm volatile("s_wait_asynccnt 0x2" ::: "memory");  // current A tile done
            if (wid == 0) __builtin_amdgcn_s_wait_tensorcnt(1); // current B tile done
#else
            asm volatile("s_wait_asynccnt 0x4" ::: "memory");  // current tile-pair done
#endif
        } else {
            asm volatile("s_wait_asynccnt 0x0" ::: "memory");
#if USE_TDM_B
            if (wid == 0) __builtin_amdgcn_s_wait_tensorcnt(0);
#endif
        }
        __syncthreads();

        Frag16 af[2];
        #pragma unroll
        for (int a = 0; a < 2; ++a) {
            const int mr = wm * 32 + a * 16 + (lane & 15);
            af[a].u[0] = *(const uint4*)&As[buf][mr][half * 8];
            af[a].u[1] = *(const uint4*)&As[buf][mr][16 + half * 8];
        }
        #pragma unroll
        for (int bb = 0; bb < 4; ++bb) {
            const int nr = wn * 64 + bb * 16 + (lane & 15);
            Frag16 bf;
            bf.u[0] = *(const uint4*)&Bs[buf][nr][half * 8];
            bf.u[1] = *(const uint4*)&Bs[buf][nr][16 + half * 8];
            #pragma unroll
            for (int a = 0; a < 2; ++a)
                acc[a][bb] = __builtin_amdgcn_wmma_f32_16x16x32_f16(
                    false, af[a].v, false, bf.v, (short)0, acc[a][bb], false, false);
        }
        __syncthreads();
    }

    #pragma unroll
    for (int a = 0; a < 2; ++a) {
        const int mbase = m0 + wm * 32 + a * 16 + (lane >> 4) * 8;
        #pragma unroll
        for (int bb = 0; bb < 4; ++bb) {
            const int nc = n0 + wn * 64 + bb * 16 + (lane & 15);
            #pragma unroll
            for (int r = 0; r < 8; ++r) {
                size_t idx = (size_t)(mbase + r) * N + nc;
                float v = acc[a][bb][r];
                if (MODE == 1) {
                    outH[idx] = (_Float16)v;
                } else if (MODE == 2) {
                    outF[idx] = resid[idx] + v * scale[nc];
                } else {
                    float u = (float)gate[idx];
                    outH[idx] = (_Float16)((v / (1.f + __expf(-v))) * u);
                }
            }
        }
    }
}

// ------------------------------------------- WMMA flash attention (windowed ALiBi)
// qkv rows: [q(1024) | k(512) | v(512)].  Mask: 0 <= j-i <= WINDOW.  GQA kv = h/2.
// Wave handles 16 queries x 64 dims; 32-key tiles; QK^T and PV on WMMA;
// online softmax in fp32 in the C-register layout; P and V^T staged in wave-private LDS.
__global__ void __launch_bounds__(256)
k_attn(const _Float16* __restrict__ qkv, _Float16* __restrict__ out) {
    __shared__ _Float16 Pbuf[8][16][32];
    __shared__ _Float16 Vt[8][64][40];
    const int b = blockIdx.z, h = blockIdx.y;
    const int w = threadIdx.x >> 5, lane = threadIdx.x & 31;
    const int i0 = (blockIdx.x * 8 + w) * 16;        // 16 queries per wave
    const int half = lane >> 4;
    const int ln15 = lane & 15;
    const int kvh = h >> 1;
    const float slope = exp2f(-0.5f * (float)(h + 1));

    // Q fragments (A layout): lane holds query row i0 + (lane&15)
    const _Float16* qrow = qkv + (size_t)(b * SEQ + i0 + ln15) * 2048 + h * HD;
    Frag16 aq0, aq1;
    aq0.u[0] = *(const uint4*)(qrow + half * 8);
    aq0.u[1] = *(const uint4*)(qrow + 16 + half * 8);
    aq1.u[0] = *(const uint4*)(qrow + 32 + half * 8);
    aq1.u[1] = *(const uint4*)(qrow + 48 + half * 8);

    v8f o[4] = {};
    float mrow[8], lrow[8];
    #pragma unroll
    for (int r = 0; r < 8; ++r) { mrow[r] = -1e30f; lrow[r] = 0.f; }

    const int jstart = (i0 / 32) * 32;
    const int jlast  = min(i0 + 15 + WINDOW, SEQ - 1);
    for (int jb = jstart; jb <= jlast; jb += 32) {
        // stage V tile (32 keys x 64 dims) transposed into Vt[w][dim][key]
        #pragma unroll
        for (int it = 0; it < 8; ++it) {
            int c = lane + it * 32;                  // 0..255
            int kk = c >> 3, d0 = (c & 7) * 8;
            int jr = min(b * SEQ + jb + kk, ROWS - 1);
            const _Float16* vr = qkv + (size_t)jr * 2048 + 1536 + kvh * HD + d0;
            uint4 vv = *(const uint4*)vr;
            _Float16 tmp[8];
            *(uint4*)tmp = vv;
            #pragma unroll
            for (int e = 0; e < 8; ++e) Vt[w][d0 + e][kk] = tmp[e];
        }

        // QK^T: two 16-key sub-tiles, K rows load directly in B layout
        v8f sc[2] = {};
        #pragma unroll
        for (int t = 0; t < 2; ++t) {
            int jr = min(b * SEQ + jb + t * 16 + ln15, ROWS - 1);
            const _Float16* kr = qkv + (size_t)jr * 2048 + 1024 + kvh * HD;
            Frag16 bk0, bk1;
            bk0.u[0] = *(const uint4*)(kr + half * 8);
            bk0.u[1] = *(const uint4*)(kr + 16 + half * 8);
            bk1.u[0] = *(const uint4*)(kr + 32 + half * 8);
            bk1.u[1] = *(const uint4*)(kr + 48 + half * 8);
            sc[t] = __builtin_amdgcn_wmma_f32_16x16x32_f16(
                false, aq0.v, false, bk0.v, (short)0, sc[t], false, false);
            sc[t] = __builtin_amdgcn_wmma_f32_16x16x32_f16(
                false, aq1.v, false, bk1.v, (short)0, sc[t], false, false);
        }

        // mask + ALiBi + online softmax (C layout: row m = r + 8*half, col n = lane&15)
        const int mrowbase = 8 * half;
        #pragma unroll
        for (int r = 0; r < 8; ++r) {
            const int i = i0 + r + mrowbase;
            float s0 = sc[0][r], s1 = sc[1][r];
            int j0 = jb + ln15, j1 = j0 + 16;
            int d0 = j0 - i, d1 = j1 - i;
            s0 = (d0 >= 0 && d0 <= WINDOW && j0 < SEQ) ? s0 * 0.125f - slope * (float)d0 : -1e30f;
            s1 = (d1 >= 0 && d1 <= WINDOW && j1 < SEQ) ? s1 * 0.125f - slope * (float)d1 : -1e30f;
            float rm = fmaxf(s0, s1);
            #pragma unroll
            for (int off = 1; off < 16; off <<= 1) rm = fmaxf(rm, __shfl_xor(rm, off, 32));
            float mn  = fmaxf(mrow[r], rm);
            float cor = __expf(mrow[r] - mn);
            mrow[r] = mn;
            float p0 = __expf(s0 - mn);
            float p1 = __expf(s1 - mn);
            float ps = p0 + p1;
            #pragma unroll
            for (int off = 1; off < 16; off <<= 1) ps += __shfl_xor(ps, off, 32);
            lrow[r] = lrow[r] * cor + ps;
            #pragma unroll
            for (int t2 = 0; t2 < 4; ++t2) o[t2][r] *= cor;
            Pbuf[w][r + mrowbase][ln15]      = (_Float16)p0;
            Pbuf[w][r + mrowbase][16 + ln15] = (_Float16)p1;
        }

        // PV: reload P as A fragment, V^T columns as B fragments (wave-private LDS,
        // DS ops are in-order within a wave)
        Frag16 ap;
        ap.u[0] = *(const uint4*)&Pbuf[w][ln15][half * 8];
        ap.u[1] = *(const uint4*)&Pbuf[w][ln15][16 + half * 8];
        #pragma unroll
        for (int t2 = 0; t2 < 4; ++t2) {
            Frag16 bv;
            bv.u[0] = *(const uint4*)&Vt[w][t2 * 16 + ln15][half * 8];
            bv.u[1] = *(const uint4*)&Vt[w][t2 * 16 + ln15][16 + half * 8];
            o[t2] = __builtin_amdgcn_wmma_f32_16x16x32_f16(
                false, ap.v, false, bv.v, (short)0, o[t2], false, false);
        }
    }

    // normalize + store: out[(b*SEQ+i)*1024 + h*64 + d]
    #pragma unroll
    for (int r = 0; r < 8; ++r) {
        const int i = i0 + r + 8 * half;
        float inv = 1.f / lrow[r];
        _Float16* orow = out + (size_t)(b * SEQ + i) * 1024 + h * HD;
        #pragma unroll
        for (int t2 = 0; t2 < 4; ++t2)
            orow[t2 * 16 + ln15] = (_Float16)(o[t2][r] * inv);
    }
}

// ------------------------------------------------------------------- driver

extern "C" void kernel_launch(void* const* d_in, const int* in_sizes, int n_in,
                              void* d_out, int out_size, void* d_ws, size_t ws_size,
                              hipStream_t stream) {
    (void)in_sizes; (void)n_in; (void)out_size; (void)ws_size;
    const float* x           = (const float*)d_in[0];
    const float* wq          = (const float*)d_in[1];
    const float* wk          = (const float*)d_in[2];
    const float* wv          = (const float*)d_in[3];
    const float* wo          = (const float*)d_in[4];
    const float* q_norm_w    = (const float*)d_in[5];
    const float* k_norm_w    = (const float*)d_in[6];
    const float* attn_norm_w = (const float*)d_in[7];
    const float* ffn_norm_w  = (const float*)d_in[8];
    const float* w1          = (const float*)d_in[9];
    const float* w2          = (const float*)d_in[10];
    const float* w3          = (const float*)d_in[11];
    const float* attn_scale  = (const float*)d_in[12];
    const float* ffn_scale   = (const float*)d_in[13];
    float* out = (float*)d_out;

    char* wsp = (char*)d_ws;
    size_t off = 0;
    auto alloc = [&](size_t bytes) -> void* {
        void* p = wsp + off;
        off = (off + bytes + 255) & ~(size_t)255;
        return p;
    };
    _Float16* wqkv_h = (_Float16*)alloc((size_t)2048 * 1024 * 2);
    _Float16* wo_h   = (_Float16*)alloc((size_t)1024 * 1024 * 2);
    _Float16* w1_h   = (_Float16*)alloc((size_t)HIDDEN * 1024 * 2);
    _Float16* w3_h   = (_Float16*)alloc((size_t)HIDDEN * 1024 * 2);
    _Float16* w2_h   = (_Float16*)alloc((size_t)1024 * HIDDEN * 2);
    _Float16* h_h    = (_Float16*)alloc((size_t)ROWS * 1024 * 2);
    _Float16* qkv_h  = (_Float16*)alloc((size_t)ROWS * 2048 * 2);
    _Float16* ao_h   = (_Float16*)alloc((size_t)ROWS * 1024 * 2);
    float*    x1     = (float*)   alloc((size_t)ROWS * 1024 * 4);
    _Float16* h2_h   = (_Float16*)alloc((size_t)ROWS * 1024 * 2);
    _Float16* u3_h   = (_Float16*)alloc((size_t)ROWS * HIDDEN * 2);
    _Float16* g_h    = (_Float16*)alloc((size_t)ROWS * HIDDEN * 2);

    // 1. weight conversion (wq|wk|wv packed to [2048,1024])
    k_cvt_f16<<<2048, 256, 0, stream>>>(wq, wqkv_h,                1024 * 1024);
    k_cvt_f16<<<2048, 256, 0, stream>>>(wk, wqkv_h + 1024 * 1024,   512 * 1024);
    k_cvt_f16<<<2048, 256, 0, stream>>>(wv, wqkv_h + 1536 * 1024,   512 * 1024);
    k_cvt_f16<<<2048, 256, 0, stream>>>(wo, wo_h,                  1024 * 1024);
    k_cvt_f16<<<2048, 256, 0, stream>>>(w1, w1_h,                  HIDDEN * 1024);
    k_cvt_f16<<<2048, 256, 0, stream>>>(w3, w3_h,                  HIDDEN * 1024);
    k_cvt_f16<<<2048, 256, 0, stream>>>(w2, w2_h,                  1024 * HIDDEN);

    // 2. attn LayerNorm
    k_layernorm_f16<<<ROWS, 256, 0, stream>>>(x, attn_norm_w, h_h, 1024, 1e-5f);

    // 3. QKV GEMM (f16 out) + Q/K norms
    k_gemm<1><<<dim3(2048 / 128, ROWS / 128), 256, 0, stream>>>(
        h_h, wqkv_h, ROWS, 2048, 1024, nullptr, qkv_h, nullptr, nullptr, nullptr);
    k_qk_norm<<<ROWS, 256, 0, stream>>>(qkv_h, q_norm_w, k_norm_w);

    // 4. flash attention (WMMA)
    k_attn<<<dim3(SEQ / 128, NH, BATCH), 256, 0, stream>>>(qkv_h, ao_h);

    // 5. WO GEMM + residual*attn_scale -> x1
    k_gemm<2><<<dim3(1024 / 128, ROWS / 128), 256, 0, stream>>>(
        ao_h, wo_h, ROWS, 1024, 1024, x1, nullptr, x, attn_scale, nullptr);

    // 6. FFN
    k_layernorm_f16<<<ROWS, 256, 0, stream>>>(x1, ffn_norm_w, h2_h, 1024, 1e-5f);
    k_gemm<1><<<dim3(HIDDEN / 128, ROWS / 128), 256, 0, stream>>>(
        h2_h, w3_h, ROWS, HIDDEN, 1024, nullptr, u3_h, nullptr, nullptr, nullptr);
    k_gemm<3><<<dim3(HIDDEN / 128, ROWS / 128), 256, 0, stream>>>(
        h2_h, w1_h, ROWS, HIDDEN, 1024, nullptr, g_h, nullptr, nullptr, u3_h);
    k_gemm<2><<<dim3(1024 / 128, ROWS / 128), 256, 0, stream>>>(
        g_h, w2_h, ROWS, 1024, HIDDEN, out, nullptr, x1, ffn_scale, nullptr);
}
